// Seq2SeqNet_47854525612420
// MI455X (gfx1250) — matrix operand order repaired and checked by reading, not compiled, not run on hardware
//
#include <hip/hip_runtime.h>
#include <hip/hip_bf16.h>

// ---------------------------------------------------------------------------
// Types for CDNA5 WMMA (wave32): v_wmma_f32_16x16x32_bf16
// ---------------------------------------------------------------------------
typedef __attribute__((ext_vector_type(16))) __bf16 v16bf;
typedef __attribute__((ext_vector_type(8)))  __bf16 v8bf;
typedef __attribute__((ext_vector_type(4)))  __bf16 v4bf;
typedef __attribute__((ext_vector_type(8)))  float  v8f;

union FragAB { v16bf v; v8bf h[2]; };

__device__ __forceinline__ __bf16 f2bf(float f) {
  unsigned u = __builtin_bit_cast(unsigned, f);
  unsigned r = u + 0x7FFFu + ((u >> 16) & 1u);     // round-to-nearest-even
  unsigned short hs = (unsigned short)(r >> 16);
  return __builtin_bit_cast(__bf16, hs);
}
__device__ __forceinline__ float bf2f(__bf16 b) {
  unsigned short hs = __builtin_bit_cast(unsigned short, b);
  unsigned u = ((unsigned)hs) << 16;
  return __builtin_bit_cast(float, u);
}
__device__ __forceinline__ float sigm(float x) { return 1.f / (1.f + expf(-x)); }

// 32-bit LDS byte offset of a __shared__ object (for async-to-LDS ops)
__device__ __forceinline__ unsigned lds_offset(const void* p) {
  typedef __attribute__((address_space(3))) const void lds_void;
  return (unsigned)(size_t)(lds_void*)p;
}

// ---------------------------------------------------------------------------
// WMMA GEMM, NT N-tiles per wave (A fragment reused NT times):
//   C[M, n0 + grpN*NT*16) (f32) = A[M,K](bf16) * W[N,K](bf16)^T (+bias) (relu?)
// Dims multiples of 16 (K of 32). One wave = 16 rows x NT*16 cols.
// ---------------------------------------------------------------------------
template <int NT>
__global__ void wmma_gemm_bf16_nt(const __bf16* __restrict__ A, int lda,
                                  const __bf16* __restrict__ W, int ldw,
                                  float* __restrict__ C, int ldc,
                                  const float* __restrict__ bias,
                                  __bf16* __restrict__ Cbf,
                                  int M, int n0, int grpN, int K, int relu)
{
  int wave = (int)((blockIdx.x * blockDim.x + threadIdx.x) >> 5);
  int lane = threadIdx.x & 31;
  int nGroups = (M >> 4) * grpN;
  if (wave >= nGroups) return;             // wave-uniform: EXEC stays all-ones
  int gm = wave / grpN;
  int gn = wave - gm * grpN;

  // A fragment: lane l -> row gm*16+(l&15); K chunks [kb,kb+8),[kb+16,kb+24), kb=(l>>4)*8
  const __bf16* pa  = A + (size_t)(gm * 16 + (lane & 15)) * lda + ((lane >> 4) << 3);
  // B fragment: lane l -> col +(l&15); K chunks [kb,kb+8),[kb+16,kb+24), kb=(l>>4)*8
  const __bf16* pb0 = W + (size_t)(n0 + gn * (NT * 16) + (lane & 15)) * ldw
                        + ((lane >> 4) << 3);

  v8f acc[NT];
#pragma unroll
  for (int i = 0; i < NT; ++i) {
    v8f z = {0.f,0.f,0.f,0.f,0.f,0.f,0.f,0.f};
    acc[i] = z;
  }

  for (int k = 0; k < K; k += 32) {
    FragAB a;
    a.h[0] = *(const v8bf*)(pa + k);
    a.h[1] = *(const v8bf*)(pa + k + 16);
#pragma unroll
    for (int i = 0; i < NT; ++i) {
      FragAB b;
      const __bf16* pb = pb0 + (size_t)i * 16 * ldw + k;
      b.h[0] = *(const v8bf*)(pb);
      b.h[1] = *(const v8bf*)(pb + 16);
      acc[i] = __builtin_amdgcn_wmma_f32_16x16x32_bf16(false, a.v, false, b.v,
                                                       (short)0, acc[i], false, false);
    }
  }

  int mb = gm * 16 + ((lane >> 4) << 3);
#pragma unroll
  for (int i = 0; i < NT; ++i) {
    int n = n0 + (gn * NT + i) * 16 + (lane & 15);
    float bv = bias ? bias[n] : 0.f;
#pragma unroll
    for (int r = 0; r < 8; ++r) {
      float v = acc[i][r] + bv;
      if (relu) v = fmaxf(v, 0.f);
      size_t idx = (size_t)(mb + r) * ldc + n;
      C[idx] = v;
      if (Cbf) Cbf[idx] = f2bf(v);
    }
  }
}

// ---------------------------------------------------------------------------
// Prologue helpers
// ---------------------------------------------------------------------------
__global__ void cast_bf16(const float* __restrict__ src, __bf16* __restrict__ dst, int n)
{
  int i = blockIdx.x * blockDim.x + threadIdx.x;
  if (i < n) dst[i] = f2bf(src[i]);
}

// xs[(t*64+b)*512 + h] = emb_jamo[input_tokens[b][t]][h]  (bf16)
__global__ void gather_embed(const int* __restrict__ toks, const float* __restrict__ emb,
                             __bf16* __restrict__ xs)
{
  size_t i = (size_t)blockIdx.x * blockDim.x + threadIdx.x;
  if (i >= (size_t)64 * 1024 * 512) return;
  int h = (int)(i & 511);
  size_t m = i >> 9;
  int b = (int)(m & 63);
  int t = (int)(m >> 6);
  xs[i] = f2bf(emb[(size_t)toks[b * 1024 + t] * 512 + h]);
}

__global__ void init_tok(int* tok) { if (threadIdx.x < 64) tok[threadIdx.x] = 1; /* <s> */ }

// ---------------------------------------------------------------------------
// Encoder scan: persistent single workgroup (1024 threads = 32 waves).
// LDS: h_f(bf16 64x256) + h_b(bf16 64x256) + gh(f32 64x768) = 256 KB dynamic.
// Per step & direction: WMMA gh = h @ Whh^T (M=64,N=768,K=256) with NT=2
// register blocking (96 groups = 3 per wave), then GRU gates.
// h kept in f32 registers (16 elems/thread/dir) for full-precision recurrence.
// ---------------------------------------------------------------------------
__global__ __launch_bounds__(1024)
void encoder_scan(const float* __restrict__ gi_f, const float* __restrict__ gi_b,
                  const __bf16* __restrict__ Whh_f, const __bf16* __restrict__ Whh_b,
                  const float* __restrict__ bhh_f, const float* __restrict__ bhh_b,
                  __bf16* __restrict__ enc_out,      // [64][1024][512] bf16
                  float* __restrict__ dec_h,         // [64][512] f32 (h0 for decoder)
                  __bf16* __restrict__ dec_h_bf)     // [64][512] bf16
{
  extern __shared__ char smem[];
  __bf16* hsh[2];
  hsh[0] = (__bf16*)smem;            // h_f  [64][256]
  hsh[1] = hsh[0] + 64 * 256;        // h_b  [64][256]
  float* gh = (float*)(smem + (size_t)64 * 256 * 2 * sizeof(__bf16)); // [64][768]

  const int tid  = threadIdx.x;
  const int lane = tid & 31;
  const int wave = tid >> 5;
  const int j    = tid & 255;        // hidden index owned by this thread
  const int brow = tid >> 8;         // b = brow + 4*q

  float hreg[2][16];
#pragma unroll
  for (int d = 0; d < 2; ++d)
#pragma unroll
    for (int q = 0; q < 16; ++q) {
      hreg[d][q] = 0.f;
      hsh[d][((brow + 4 * q) << 8) + j] = f2bf(0.f);
    }
  __syncthreads();

  for (int t = 0; t < 1024; ++t) {
#pragma unroll
    for (int d = 0; d < 2; ++d) {
      const __bf16* Whh = d ? Whh_b : Whh_f;
      const float*  gi  = d ? gi_b  : gi_f;
      const float*  bhh = d ? bhh_b : bhh_f;
      const __bf16* H   = hsh[d];

      // gh = H @ Whh^T : 4 M-tiles x 24 N-groups (NT=2) = 96 groups, 3/wave
      for (int grp = wave; grp < 96; grp += 32) {
        int gm = grp / 24, gn = grp - gm * 24;
        const __bf16* pa  = H + ((gm * 16 + (lane & 15)) << 8) + ((lane >> 4) << 3);
        const __bf16* pb0 = Whh + ((size_t)(gn * 32 + (lane & 15)) << 8)
                                + ((lane >> 4) << 3);
        v8f acc0 = {0.f,0.f,0.f,0.f,0.f,0.f,0.f,0.f};
        v8f acc1 = {0.f,0.f,0.f,0.f,0.f,0.f,0.f,0.f};
#pragma unroll
        for (int k = 0; k < 256; k += 32) {
          FragAB a, b0, b1;
          a.h[0]  = *(const v8bf*)(pa + k);
          a.h[1]  = *(const v8bf*)(pa + k + 16);
          b0.h[0] = *(const v8bf*)(pb0 + k);
          b0.h[1] = *(const v8bf*)(pb0 + k + 16);
          b1.h[0] = *(const v8bf*)(pb0 + (16 << 8) + k);
          b1.h[1] = *(const v8bf*)(pb0 + (16 << 8) + k + 16);
          acc0 = __builtin_amdgcn_wmma_f32_16x16x32_bf16(false, a.v, false, b0.v,
                                                         (short)0, acc0, false, false);
          acc1 = __builtin_amdgcn_wmma_f32_16x16x32_bf16(false, a.v, false, b1.v,
                                                         (short)0, acc1, false, false);
        }
        int n  = gn * 32 + (lane & 15);
        int mb = gm * 16 + ((lane >> 4) << 3);
#pragma unroll
        for (int r = 0; r < 8; ++r) {
          gh[(mb + r) * 768 + n]      = acc0[r];
          gh[(mb + r) * 768 + n + 16] = acc1[r];
        }
      }
      __syncthreads();

      // GRU gates (r,z,n ordering as PyTorch)
#pragma unroll 4
      for (int q = 0; q < 16; ++q) {
        int b = brow + 4 * q;
        const float* gib = gi + ((size_t)t * 64 + b) * 768;   // includes bih
        const float* ghb = gh + b * 768;                       // excludes bhh
        float rr = sigm(gib[j]       + ghb[j]       + bhh[j]);
        float zz = sigm(gib[256 + j] + ghb[256 + j] + bhh[256 + j]);
        float nn = tanhf(gib[512 + j] + rr * (ghb[512 + j] + bhh[512 + j]));
        float hv = (1.f - zz) * nn + zz * hreg[d][q];
        hreg[d][q] = hv;
        hsh[d][(b << 8) + j] = f2bf(hv);
        enc_out[(((size_t)b << 10) + t) * 512 + (d << 8) + j] = f2bf(hv);
        if (t == 1023) {
          dec_h   [(b << 9) + (d << 8) + j] = hv;
          dec_h_bf[(b << 9) + (d << 8) + j] = f2bf(hv);
        }
      }
      __syncthreads();
    }
  }
}

// ---------------------------------------------------------------------------
// Decoder step kernels
// ---------------------------------------------------------------------------
// A1[b][0..511]=emb_char_bf[tok[b]], A1[b][512..1023]=h (bf16)
__global__ void build_A1(const int* __restrict__ tok, const __bf16* __restrict__ embc,
                         const __bf16* __restrict__ hbf, __bf16* __restrict__ A1)
{
  int b = blockIdx.x, i = threadIdx.x;   // 1024 threads
  A1[(b << 10) + i] = (i < 512) ? embc[((size_t)tok[b] << 9) + i]
                                : hbf[(b << 9) + (i - 512)];
}

// scores[b][l] = enc_out[b][l][:] . key[b][:]   (one wave per (b,l), 512 dot)
__global__ void attn_scores(const __bf16* __restrict__ enc, const float* __restrict__ key,
                            float* __restrict__ scores)
{
  int gw = (int)((blockIdx.x * blockDim.x + threadIdx.x) >> 5);
  int lane = threadIdx.x & 31;
  if (gw >= 64 * 1024) return;
  int b = gw >> 10, l = gw & 1023;
  int h0 = lane << 4;
  const __bf16* e  = enc + ((size_t)((b << 10) + l) << 9) + h0;
  const float*  kk = key + (b << 9) + h0;
  v8bf e0 = *(const v8bf*)(e);
  v8bf e1 = *(const v8bf*)(e + 8);
  float s = 0.f;
#pragma unroll
  for (int r = 0; r < 8; ++r) s += bf2f(e0[r]) * kk[r];
#pragma unroll
  for (int r = 0; r < 8; ++r) s += bf2f(e1[r]) * kk[8 + r];
#pragma unroll
  for (int off = 16; off > 0; off >>= 1) s += __shfl_down(s, off, 32);
  if (lane == 0) scores[(b << 10) + l] = s;
}

__global__ void attn_softmax(const float* __restrict__ scores, float* __restrict__ w,
                             float* __restrict__ attn_out, int t)
{
  __shared__ float red[256];
  int b = blockIdx.x, tid = threadIdx.x;
  const float* sc = scores + (b << 10);
  float m = -3.4e38f;
  for (int l = tid; l < 1024; l += 256) m = fmaxf(m, sc[l]);
  red[tid] = m; __syncthreads();
  for (int st = 128; st > 0; st >>= 1) {
    if (tid < st) red[tid] = fmaxf(red[tid], red[tid + st]);
    __syncthreads();
  }
  float mx = red[0]; __syncthreads();
  float sum = 0.f;
  for (int l = tid; l < 1024; l += 256) sum += expf(sc[l] - mx);
  red[tid] = sum; __syncthreads();
  for (int st = 128; st > 0; st >>= 1) {
    if (tid < st) red[tid] += red[tid + st];
    __syncthreads();
  }
  float inv = 1.f / red[0];
  for (int l = tid; l < 1024; l += 256) {
    float v = expf(sc[l] - mx) * inv;
    w[(b << 10) + l] = v;
    attn_out[(size_t)((b << 10) + l) * 256 + t] = v;   // [B][Lin][T]
  }
}

// ctx[b][h] = sum_l w[b][l]*enc[b][l][h]; also builds A2 = [emb || ctx] bf16.
// Attention weights staged into LDS with CDNA5 async global->LDS copies
// (tracked by ASYNCcnt, released with s_wait_asynccnt + barrier).
__global__ void attn_ctx(const __bf16* __restrict__ enc, const float* __restrict__ w,
                         float* __restrict__ ctx, const __bf16* __restrict__ A1,
                         __bf16* __restrict__ A2)
{
  __shared__ float wl[1024];
  int b = blockIdx.x, tid = threadIdx.x;    // 128 threads, 4 h-values each
  {
    unsigned loff = lds_offset(wl + tid * 4);               // 16B per async op
    unsigned long long g0 = (unsigned long long)(size_t)(w + (b << 10) + tid * 4);
    unsigned long long g1 = g0 + 512 * sizeof(float);
    asm volatile("global_load_async_to_lds_b128 %0, %1, off"
                 :: "v"(loff), "v"(g0) : "memory");
    asm volatile("global_load_async_to_lds_b128 %0, %1, off"
                 :: "v"(loff + 2048u), "v"(g1) : "memory");
    asm volatile("s_wait_asynccnt 0x0" ::: "memory");
  }
  __syncthreads();

  int h0 = tid * 4;
  const __bf16* e = enc + ((size_t)b << 19) + h0;
  float a0 = 0.f, a1 = 0.f, a2 = 0.f, a3 = 0.f;
  for (int l = 0; l < 1024; ++l) {
    v4bf ev = *(const v4bf*)(e + ((size_t)l << 9));
    float wv = wl[l];
    a0 += wv * bf2f(ev[0]);
    a1 += wv * bf2f(ev[1]);
    a2 += wv * bf2f(ev[2]);
    a3 += wv * bf2f(ev[3]);
  }
  ctx[(b << 9) + h0 + 0] = a0;
  ctx[(b << 9) + h0 + 1] = a1;
  ctx[(b << 9) + h0 + 2] = a2;
  ctx[(b << 9) + h0 + 3] = a3;
#pragma unroll
  for (int i = 0; i < 4; ++i)                       // emb half copy (512 = 128*4)
    A2[(b << 10) + h0 + i] = A1[(b << 10) + h0 + i];
  A2[(b << 10) + 512 + h0 + 0] = f2bf(a0);
  A2[(b << 10) + 512 + h0 + 1] = f2bf(a1);
  A2[(b << 10) + 512 + h0 + 2] = f2bf(a2);
  A2[(b << 10) + 512 + h0 + 3] = f2bf(a3);
}

// GRU gate update for decoder (gi/gh already include biases from GEMM)
__global__ void dec_gates(const float* __restrict__ gi, const float* __restrict__ ghm,
                          float* __restrict__ h, __bf16* __restrict__ hbf)
{
  int i = blockIdx.x * blockDim.x + threadIdx.x;   // 64*512
  int b = i >> 9, j = i & 511;
  const float* a = gi + b * 1536;
  const float* c = ghm + b * 1536;
  float rr = sigm(a[j] + c[j]);
  float zz = sigm(a[512 + j] + c[512 + j]);
  float nn = tanhf(a[1024 + j] + rr * c[1024 + j]);
  float hv = (1.f - zz) * nn + zz * h[i];
  h[i] = hv;
  hbf[i] = f2bf(hv);
}

// log-softmax over VC=2000 + first-index argmax + NLL accumulation
__global__ void out_logsoftmax(const float* __restrict__ logits,
                               const int* __restrict__ target, const float* __restrict__ mask,
                               float* __restrict__ logp_out, int t,
                               int* __restrict__ tok, float* __restrict__ nll_tb)
{
  __shared__ float sv[256];
  __shared__ int   si[256];
  int b = blockIdx.x, tid = threadIdx.x;
  const float* lg = logits + b * 2000;
  float best = -3.4e38f; int bi = 0;
  for (int v = tid; v < 2000; v += 256) {
    float x = lg[v];
    if (x > best) { best = x; bi = v; }
  }
  sv[tid] = best; si[tid] = bi; __syncthreads();
  for (int st = 128; st > 0; st >>= 1) {
    if (tid < st) {
      float ov = sv[tid + st]; int oi = si[tid + st];
      if (ov > sv[tid] || (ov == sv[tid] && oi < si[tid])) { sv[tid] = ov; si[tid] = oi; }
    }
    __syncthreads();
  }
  float mx = sv[0]; int amax = si[0]; __syncthreads();
  float sum = 0.f;
  for (int v = tid; v < 2000; v += 256) sum += expf(lg[v] - mx);
  sv[tid] = sum; __syncthreads();
  for (int st = 128; st > 0; st >>= 1) {
    if (tid < st) sv[tid] += sv[tid + st];
    __syncthreads();
  }
  float lse = mx + logf(sv[0]);
  float* dst = logp_out + ((size_t)b * 256 + t) * 2000;   // [B][T][VC]
  for (int v = tid; v < 2000; v += 256) dst[v] = lg[v] - lse;
  if (tid == 0) {
    tok[b] = amax;
    int tg = target[b * 256 + t];
    nll_tb[t * 64 + b] = -(lg[tg] - lse) * mask[b * 256 + t];
  }
}

// loss = (1/T) * sum_t mean_b(nll_tb[t][b])   (fixed-order, deterministic)
__global__ void loss_reduce(const float* __restrict__ nll_tb, float* __restrict__ loss)
{
  __shared__ float s[256];
  int t = threadIdx.x;
  float a = 0.f;
  for (int b = 0; b < 64; ++b) a += nll_tb[t * 64 + b];
  s[t] = a * (1.f / 64.f);
  __syncthreads();
  for (int st = 128; st > 0; st >>= 1) {
    if (t < st) s[t] += s[t + st];
    __syncthreads();
  }
  if (t == 0) loss[0] = s[0] * (1.f / 256.f);
}

// ---------------------------------------------------------------------------
// Host side
// ---------------------------------------------------------------------------
extern "C" void kernel_launch(void* const* d_in, const int* in_sizes, int n_in,
                              void* d_out, int out_size, void* d_ws, size_t ws_size,
                              hipStream_t stream)
{
  (void)in_sizes; (void)n_in; (void)out_size; (void)ws_size;

  const int*   input_tokens  = (const int*)d_in[0];
  const int*   target_tokens = (const int*)d_in[1];
  const float* loss_mask     = (const float*)d_in[2];
  const float* emb_jamo      = (const float*)d_in[3];
  const float* emb_char      = (const float*)d_in[4];
  const float* enc_Wih_f     = (const float*)d_in[5];
  const float* enc_Whh_f     = (const float*)d_in[6];
  const float* enc_bih_f     = (const float*)d_in[7];
  const float* enc_bhh_f     = (const float*)d_in[8];
  const float* enc_Wih_b     = (const float*)d_in[9];
  const float* enc_Whh_b     = (const float*)d_in[10];
  const float* enc_bih_b     = (const float*)d_in[11];
  const float* enc_bhh_b     = (const float*)d_in[12];
  const float* dec_Wih       = (const float*)d_in[13];
  const float* dec_Whh       = (const float*)d_in[14];
  const float* dec_bih       = (const float*)d_in[15];
  const float* dec_bhh       = (const float*)d_in[16];
  const float* attn_W        = (const float*)d_in[17];
  const float* attn_b        = (const float*)d_in[18];
  const float* comb_W        = (const float*)d_in[19];
  const float* comb_b        = (const float*)d_in[20];
  const float* out_W         = (const float*)d_in[21];
  const float* out_b         = (const float*)d_in[22];

  char* ws = (char*)d_ws;
  size_t off = 0;
  auto alloc = [&](size_t bytes) { size_t o = off; off += (bytes + 255) & ~(size_t)255; return o; };

  __bf16* enc_Wih_f_bf = (__bf16*)(ws + alloc((size_t)768 * 512 * 2));
  __bf16* enc_Whh_f_bf = (__bf16*)(ws + alloc((size_t)768 * 256 * 2));
  __bf16* enc_Wih_b_bf = (__bf16*)(ws + alloc((size_t)768 * 512 * 2));
  __bf16* enc_Whh_b_bf = (__bf16*)(ws + alloc((size_t)768 * 256 * 2));
  __bf16* dec_Wih_bf   = (__bf16*)(ws + alloc((size_t)1536 * 512 * 2));
  __bf16* dec_Whh_bf   = (__bf16*)(ws + alloc((size_t)1536 * 512 * 2));
  __bf16* attn_W_bf    = (__bf16*)(ws + alloc((size_t)512 * 1024 * 2));
  __bf16* comb_W_bf    = (__bf16*)(ws + alloc((size_t)512 * 1024 * 2));
  __bf16* out_W_bf     = (__bf16*)(ws + alloc((size_t)2000 * 512 * 2));
  __bf16* emb_char_bf  = (__bf16*)(ws + alloc((size_t)2000 * 512 * 2));
  __bf16* xs_bf        = (__bf16*)(ws + alloc((size_t)64 * 1024 * 512 * 2));
  __bf16* enc_out_bf   = (__bf16*)(ws + alloc((size_t)64 * 1024 * 512 * 2));
  float*  gi_f         = (float*)(ws + alloc((size_t)65536 * 768 * 4));
  float*  gi_b         = (float*)(ws + alloc((size_t)65536 * 768 * 4));
  float*  dec_h        = (float*)(ws + alloc((size_t)64 * 512 * 4));
  __bf16* dec_h_bf     = (__bf16*)(ws + alloc((size_t)64 * 512 * 2));
  __bf16* A1           = (__bf16*)(ws + alloc((size_t)64 * 1024 * 2));
  __bf16* A2           = (__bf16*)(ws + alloc((size_t)64 * 1024 * 2));
  float*  keyb         = (float*)(ws + alloc((size_t)64 * 512 * 4));
  float*  scores       = (float*)(ws + alloc((size_t)64 * 1024 * 4));
  float*  wbuf         = (float*)(ws + alloc((size_t)64 * 1024 * 4));
  float*  ctx          = (float*)(ws + alloc((size_t)64 * 512 * 4));
  float*  gbuf         = (float*)(ws + alloc((size_t)64 * 512 * 4));
  __bf16* g_bf         = (__bf16*)(ws + alloc((size_t)64 * 512 * 2));
  float*  gates_i      = (float*)(ws + alloc((size_t)64 * 1536 * 4));
  float*  gates_h      = (float*)(ws + alloc((size_t)64 * 1536 * 4));
  float*  logits       = (float*)(ws + alloc((size_t)64 * 2000 * 4));
  float*  nll_tb       = (float*)(ws + alloc((size_t)256 * 64 * 4));
  int*    tok          = (int*)(ws + alloc((size_t)64 * 4));

  float* out_f     = (float*)d_out;
  float* attn_base = out_f + (size_t)64 * 256 * 2000;
  float* loss_ptr  = attn_base + (size_t)64 * 1024 * 256;

  auto cast = [&](const float* s, __bf16* d, int n) {
    cast_bf16<<<(n + 255) / 256, 256, 0, stream>>>(s, d, n);
  };
  // NT=4 register-blocked GEMM with NT=1 remainder launch (N=2000 -> 124+1 tiles)
  auto gemm = [&](const __bf16* A, int lda, const __bf16* W, int ldw, float* C, int ldc,
                  const float* bias, __bf16* Cbf, int M, int N, int K, int relu) {
    int tilesN = N / 16;
    int grp4 = tilesN / 4;
    if (grp4 > 0) {
      int groups = (M / 16) * grp4;
      wmma_gemm_bf16_nt<4><<<(groups + 7) / 8, 256, 0, stream>>>(
          A, lda, W, ldw, C, ldc, bias, Cbf, M, 0, grp4, K, relu);
    }
    int rem = tilesN - grp4 * 4;
    if (rem > 0) {
      int groups = (M / 16) * rem;
      wmma_gemm_bf16_nt<1><<<(groups + 7) / 8, 256, 0, stream>>>(
          A, lda, W, ldw, C, ldc, bias, Cbf, M, grp4 * 64, rem, K, relu);
    }
  };

  // --- Prologue: weight casts, embedding gather, batched input projections ---
  cast(enc_Wih_f, enc_Wih_f_bf, 768 * 512);
  cast(enc_Whh_f, enc_Whh_f_bf, 768 * 256);
  cast(enc_Wih_b, enc_Wih_b_bf, 768 * 512);
  cast(enc_Whh_b, enc_Whh_b_bf, 768 * 256);
  cast(dec_Wih,   dec_Wih_bf,   1536 * 512);
  cast(dec_Whh,   dec_Whh_bf,   1536 * 512);
  cast(attn_W,    attn_W_bf,    512 * 1024);
  cast(comb_W,    comb_W_bf,    512 * 1024);
  cast(out_W,     out_W_bf,     2000 * 512);
  cast(emb_char,  emb_char_bf,  2000 * 512);

  gather_embed<<<(64 * 1024 * 512) / 256, 256, 0, stream>>>(input_tokens, emb_jamo, xs_bf);

  // gi = xs @ Wih^T + bih   (M=65536, N=768, K=512) for both directions
  gemm(xs_bf, 512, enc_Wih_f_bf, 512, gi_f, 768, enc_bih_f, nullptr, 65536, 768, 512, 0);
  gemm(xs_bf, 512, enc_Wih_b_bf, 512, gi_b, 768, enc_bih_b, nullptr, 65536, 768, 512, 0);

  init_tok<<<1, 64, 0, stream>>>(tok);

  // --- Encoder recurrence: persistent single workgroup, 256 KB dynamic LDS ---
  (void)hipFuncSetAttribute((const void*)encoder_scan,
                            hipFuncAttributeMaxDynamicSharedMemorySize, 262144);
  encoder_scan<<<1, 1024, 262144, stream>>>(gi_f, gi_b, enc_Whh_f_bf, enc_Whh_b_bf,
                                            enc_bhh_f, enc_bhh_b,
                                            enc_out_bf, dec_h, dec_h_bf);

  // --- Decoder: 256 sequential steps, stream-ordered kernels ---
  for (int t = 0; t < 256; ++t) {
    build_A1<<<64, 1024, 0, stream>>>(tok, emb_char_bf, dec_h_bf, A1);
    // key = [emb||h] @ attn_W^T + attn_b
    gemm(A1, 1024, attn_W_bf, 1024, keyb, 512, attn_b, nullptr, 64, 512, 1024, 0);
    attn_scores<<<(64 * 1024) / 8, 256, 0, stream>>>(enc_out_bf, keyb, scores);
    attn_softmax<<<64, 256, 0, stream>>>(scores, wbuf, attn_base, t);
    attn_ctx<<<64, 128, 0, stream>>>(enc_out_bf, wbuf, ctx, A1, A2);
    // g = relu([emb||ctx] @ comb_W^T + comb_b)  (also bf16 mirror for next GEMM)
    gemm(A2, 1024, comb_W_bf, 1024, gbuf, 512, comb_b, g_bf, 64, 512, 1024, 1);
    // GRU projections (biases folded into GEMM)
    gemm(g_bf,     512, dec_Wih_bf, 512, gates_i, 1536, dec_bih, nullptr, 64, 1536, 512, 0);
    gemm(dec_h_bf, 512, dec_Whh_bf, 512, gates_h, 1536, dec_bhh, nullptr, 64, 1536, 512, 0);
    dec_gates<<<128, 256, 0, stream>>>(gates_i, gates_h, dec_h, dec_h_bf);
    // output projection + log-softmax + argmax feedback
    gemm(dec_h_bf, 512, out_W_bf, 512, logits, 2000, out_b, nullptr, 64, 2000, 512, 0);
    out_logsoftmax<<<64, 256, 0, stream>>>(logits, target_tokens, loss_mask,
                                           out_f, t, tok, nll_tb);
  }

  loss_reduce<<<1, 256, 0, stream>>>(nll_tb, loss_ptr);
}
